// CTM_68968584839566
// MI455X (gfx1250) — compile-verified, matrix operand order checked by reading, not run on hardware
//
#include <hip/hip_runtime.h>
#include <hip/hip_bf16.h>
#include <math.h>

#define D_    768
#define M_    25
#define H_    4
#define W2_   16
#define SO_   384
#define BT_   2048
#define ITER_ 20
#define K1_   1536   // 2*D (concat of kv, act)

typedef __attribute__((ext_vector_type(16))) __bf16 v16bf;
typedef __attribute__((ext_vector_type(8)))  __bf16 v8bf;
typedef __attribute__((ext_vector_type(8)))  float  v8f;

__device__ __forceinline__ float sigm(float x) { return 1.f / (1.f + expf(-x)); }
__device__ __forceinline__ float silu(float x) { return x * sigm(x); }

// ---------------- block reductions (blockDim.x == 256) ----------------
__device__ __forceinline__ float2 blockReduceSum2(float2 v, float2* s) {
    int t = threadIdx.x;
    s[t] = v; __syncthreads();
    #pragma unroll
    for (int off = 128; off > 0; off >>= 1) {
        if (t < off) { s[t].x += s[t + off].x; s[t].y += s[t + off].y; }
        __syncthreads();
    }
    float2 r = s[0]; __syncthreads(); return r;
}
__device__ __forceinline__ float blockReduceSum(float v, float* s) {
    int t = threadIdx.x;
    s[t] = v; __syncthreads();
    #pragma unroll
    for (int off = 128; off > 0; off >>= 1) {
        if (t < off) s[t] += s[t + off];
        __syncthreads();
    }
    float r = s[0]; __syncthreads(); return r;
}
__device__ __forceinline__ float blockReduceMax(float v, float* s) {
    int t = threadIdx.x;
    s[t] = v; __syncthreads();
    #pragma unroll
    for (int off = 128; off > 0; off >>= 1) {
        if (t < off) s[t] = fmaxf(s[t], s[t + off]);
        __syncthreads();
    }
    float r = s[0]; __syncthreads(); return r;
}

// ---------------- one-time: repack row-major (K x N) f32 weights into WMMA
// B-fragment order: [(kb*N/16 + nb)*32 + lane]*16 + e, bf16 ----------------
__global__ void repack_w(const float* __restrict__ W, __bf16* __restrict__ Wp,
                         int K, int N) {
    int id = blockIdx.x * 256 + threadIdx.x;
    if (id >= K * N) return;
    int e    = id & 15;
    int lane = (id >> 4) & 31;
    int rest = id >> 9;
    int nb16 = N >> 4;
    int nb   = rest % nb16;
    int kb   = rest / nb16;
    int kbase = (lane >> 4) * 8;
    int k = kb * 32 + ((e < 8) ? (kbase + e) : (16 + kbase + (e - 8)));
    int n = nb * 16 + (lane & 15);
    Wp[id] = (__bf16)W[(size_t)k * N + n];
}

// ---------------- init kernels ----------------
__global__ void init_kv_act(const float* __restrict__ kv,
                            const float* __restrict__ start_act,
                            float* __restrict__ act, __bf16* __restrict__ xb) {
    int id = blockIdx.x * 256 + threadIdx.x;
    if (id >= BT_ * D_) return;
    int b = id / D_, d = id % D_;
    float a = start_act[d];
    act[id] = a;
    xb[(size_t)b * K1_ + d]       = (__bf16)kv[id];   // constant kv half
    xb[(size_t)b * K1_ + D_ + d]  = (__bf16)a;        // act half
}

__global__ void init_trace(const float* __restrict__ start_trace,
                           float* __restrict__ trace) {
    size_t id = (size_t)blockIdx.x * 256 + threadIdx.x;
    if (id >= (size_t)M_ * BT_ * D_) return;
    int m = (int)(id / ((size_t)BT_ * D_));
    int d = (int)(id % D_);
    trace[id] = start_trace[(size_t)d * M_ + m];      // logical m -> slot m initially
}

__global__ void init_sync(const float* __restrict__ start_act,
                          const int* __restrict__ il, const int* __restrict__ ir,
                          float* __restrict__ ao, float* __restrict__ bo) {
    int id = blockIdx.x * 256 + threadIdx.x;
    if (id >= BT_ * SO_) return;
    int j = id % SO_;
    ao[id] = start_act[il[j]] * start_act[ir[j]];
    bo[id] = 1.f;
}

// ---------------- bf16 WMMA GEMM: out(BT x N) = X(BT x K) @ W + bias -------
// 128 threads = 4 waves; each wave -> one 16x32 output strip (2 wmma accs).
__global__ void __launch_bounds__(128)
gemm_bf16(const __bf16* __restrict__ X, const __bf16* __restrict__ Wp,
          const float* __restrict__ bias, float* __restrict__ out,
          int K, int N) {
    int lane  = threadIdx.x & 31;
    int wave  = threadIdx.x >> 5;
    int tileM = blockIdx.y * 4 + wave;
    int tN0   = blockIdx.x * 2;
    int row   = tileM * 16 + (lane & 15);
    int kbase = (lane >> 4) * 8;
    const __bf16* xrow = X + (size_t)row * K + kbase;
    int nb16 = N >> 4;
    v8f acc0 = {};
    v8f acc1 = {};
    int nkb = K >> 5;
    for (int kb = 0; kb < nkb; ++kb) {
        const v8bf* pa = (const v8bf*)(xrow + kb * 32);
        v8bf lo = pa[0];       // K = kbase .. kbase+7
        v8bf hi = pa[2];       // K = 16+kbase .. 16+kbase+7
        v16bf af;
        #pragma unroll
        for (int e = 0; e < 8; ++e) { af[e] = lo[e]; af[e + 8] = hi[e]; }
        const v16bf* pb0 = (const v16bf*)(Wp + (((size_t)kb * nb16 + tN0) << 9));
        const v16bf* pb1 = (const v16bf*)(Wp + (((size_t)kb * nb16 + tN0 + 1) << 9));
        v16bf b0 = pb0[lane];
        v16bf b1 = pb1[lane];
        if (kb + 1 < nkb) __builtin_prefetch(xrow + (kb + 1) * 32, 0, 0);
        acc0 = __builtin_amdgcn_wmma_f32_16x16x32_bf16(false, af, false, b0,
                                                       (short)0, acc0, false, false);
        acc1 = __builtin_amdgcn_wmma_f32_16x16x32_bf16(false, af, false, b1,
                                                       (short)0, acc1, false, false);
    }
    int orow = tileM * 16 + (lane >> 4) * 8;
    int col0 = tN0 * 16 + (lane & 15);
    int col1 = col0 + 16;
    float bz0 = bias[col0], bz1 = bias[col1];
    #pragma unroll
    for (int r = 0; r < 8; ++r) {
        out[(size_t)(orow + r) * N + col0] = acc0[r] + bz0;
        out[(size_t)(orow + r) * N + col1] = acc1[r] + bz1;
    }
}

// ---------------- fused synapse UNet tail: LN+silu -> down -> LN16+silu ->
// up -> LN+silu -> skip LN -> write trace ring slot ----------------
__global__ void __launch_bounds__(256)
synnet_row(const float* __restrict__ hpre,
           const float* __restrict__ g1, const float* __restrict__ be1,
           const float* __restrict__ dw, const float* __restrict__ db,
           const float* __restrict__ dg, const float* __restrict__ dbe,
           const float* __restrict__ uw, const float* __restrict__ ub,
           const float* __restrict__ ug, const float* __restrict__ ube,
           const float* __restrict__ sg, const float* __restrict__ sbe,
           float* __restrict__ trace, int slot) {
    __shared__ float  hs[D_];
    __shared__ float2 red2[256];
    __shared__ float  part[256];
    __shared__ float  dv[W2_];
    __shared__ float  dsl[W2_];
    int b = blockIdx.x, t = threadIdx.x;
    const float* hr = hpre + (size_t)b * D_;

    // LN over 768 + silu -> hs
    float v[3];
    float2 a2{0.f, 0.f};
    #pragma unroll
    for (int i = 0; i < 3; ++i) {
        float x = hr[t + i * 256]; v[i] = x; a2.x += x; a2.y += x * x;
    }
    a2 = blockReduceSum2(a2, red2);
    float mean = a2.x * (1.f / D_);
    float var  = a2.y * (1.f / D_) - mean * mean;
    float rs   = rsqrtf(var + 1e-5f);
    #pragma unroll
    for (int i = 0; i < 3; ++i) {
        int c = t + i * 256;
        float xn = (v[i] - mean) * rs * g1[c] + be1[c];
        hs[c] = silu(xn);
    }
    __syncthreads();

    // down proj (768 -> 16), segmented reduction
    int j = t & 15, seg = t >> 4;
    float pj = 0.f;
    #pragma unroll 8
    for (int ii = 0; ii < 48; ++ii) {
        int ic = seg * 48 + ii;
        pj += hs[ic] * dw[ic * W2_ + j];
    }
    part[t] = pj; __syncthreads();
    if (t < W2_) {
        float s = db[t];
        #pragma unroll
        for (int k = 0; k < 16; ++k) s += part[k * 16 + t];
        dv[t] = s;
    }
    __syncthreads();

    // LN over 16 + silu (redundant per-thread stats: 16 broadcast reads)
    float m16 = 0.f;
    #pragma unroll
    for (int w = 0; w < W2_; ++w) m16 += dv[w];
    m16 *= (1.f / W2_);
    float v16 = 0.f;
    #pragma unroll
    for (int w = 0; w < W2_; ++w) { float dd = dv[w] - m16; v16 += dd * dd; }
    v16 *= (1.f / W2_);
    float rs16 = rsqrtf(v16 + 1e-5f);
    if (t < W2_) {
        float xn = (dv[t] - m16) * rs16 * dg[t] + dbe[t];
        dsl[t] = silu(xn);
    }
    __syncthreads();

    // up proj (16 -> 768) + LN + silu, then skip-add + LN -> trace slot
    float uv[3];
    a2 = float2{0.f, 0.f};
    #pragma unroll
    for (int i = 0; i < 3; ++i) {
        int c = t + i * 256;
        float u = ub[c];
        #pragma unroll
        for (int w = 0; w < W2_; ++w) u += dsl[w] * uw[w * D_ + c];
        uv[i] = u; a2.x += u; a2.y += u * u;
    }
    a2 = blockReduceSum2(a2, red2);
    mean = a2.x * (1.f / D_);
    var  = a2.y * (1.f / D_) - mean * mean;
    rs   = rsqrtf(var + 1e-5f);
    float sv[3];
    float2 a3{0.f, 0.f};
    #pragma unroll
    for (int i = 0; i < 3; ++i) {
        int c = t + i * 256;
        float xn = (uv[i] - mean) * rs * ug[c] + ube[c];
        float s  = silu(xn) + hs[c];
        sv[i] = s; a3.x += s; a3.y += s * s;
    }
    a3 = blockReduceSum2(a3, red2);
    mean = a3.x * (1.f / D_);
    var  = a3.y * (1.f / D_) - mean * mean;
    rs   = rsqrtf(var + 1e-5f);
    float* tp = trace + (size_t)slot * (BT_ * D_) + (size_t)b * D_;
    #pragma unroll
    for (int i = 0; i < 3; ++i) {
        int c = t + i * 256;
        tp[c] = (sv[i] - mean) * rs * sg[c] + sbe[c];
    }
}

// ---------------- per-neuron NLM: trace ring (M=25) -> act; also stage bf16
// act half of the GEMM input ----------------
__global__ void __launch_bounds__(256)
nlm_kernel(const float* __restrict__ trace,
           const float* __restrict__ g1, const float* __restrict__ be1,
           const float* __restrict__ w1, const float* __restrict__ b1,
           const float* __restrict__ T1p,
           const float* __restrict__ g2, const float* __restrict__ be2,
           const float* __restrict__ w2, const float* __restrict__ b2,
           const float* __restrict__ T2p,
           float* __restrict__ act, __bf16* __restrict__ xb, int t) {
    int id = blockIdx.x * 256 + threadIdx.x;
    if (id >= BT_ * D_) return;
    int b = id / D_, d = id % D_;

    float x[M_];
    float mean = 0.f;
    #pragma unroll
    for (int m = 0; m < M_; ++m) {
        int slot = (t + 1 + m) % M_;                    // logical m -> ring slot
        float xv = trace[(size_t)slot * (BT_ * D_) + id];
        x[m] = xv; mean += xv;
    }
    mean *= (1.f / M_);
    float var = 0.f;
    #pragma unroll
    for (int m = 0; m < M_; ++m) { float dd = x[m] - mean; var += dd * dd; }
    var *= (1.f / M_);
    float rs = rsqrtf(var + 1e-5f);

    float invT1 = 1.f / T1p[0];
    float o[2 * H_];
    #pragma unroll
    for (int h = 0; h < 2 * H_; ++h) o[h] = b1[(size_t)d * (2 * H_) + h];
    #pragma unroll
    for (int m = 0; m < M_; ++m) {
        float xn = (x[m] - mean) * rs * g1[m] + be1[m];
        #pragma unroll
        for (int h = 0; h < 2 * H_; ++h)
            o[h] += xn * w1[((size_t)m * (2 * H_) + h) * D_ + d];
    }
    float gl[H_];
    #pragma unroll
    for (int h = 0; h < H_; ++h) {
        float a = o[h] * invT1, g = o[h + H_] * invT1;
        gl[h] = a * sigm(g);                             // GLU
    }
    float m2 = 0.f;
    #pragma unroll
    for (int h = 0; h < H_; ++h) m2 += gl[h];
    m2 *= (1.f / H_);
    float v2 = 0.f;
    #pragma unroll
    for (int h = 0; h < H_; ++h) { float dd = gl[h] - m2; v2 += dd * dd; }
    v2 *= (1.f / H_);
    float rs2 = rsqrtf(v2 + 1e-5f);

    float invT2 = 1.f / T2p[0];
    float o2a = b2[(size_t)d * 2 + 0], o2b = b2[(size_t)d * 2 + 1];
    #pragma unroll
    for (int h = 0; h < H_; ++h) {
        float xn = (gl[h] - m2) * rs2 * g2[h] + be2[h];
        o2a += xn * w2[((size_t)h * 2 + 0) * D_ + d];
        o2b += xn * w2[((size_t)h * 2 + 1) * D_ + d];
    }
    o2a *= invT2; o2b *= invT2;
    float a = o2a * sigm(o2b);
    act[id] = a;
    xb[(size_t)b * K1_ + D_ + d] = (__bf16)a;            // refresh act half
}

// ---------------- decayed out-sync accumulators -> bf16 GEMM input --------
__global__ void outsync(const float* __restrict__ act,
                        const int* __restrict__ il, const int* __restrict__ ir,
                        const float* __restrict__ decay,
                        float* __restrict__ ao, float* __restrict__ bo,
                        __bf16* __restrict__ syncb) {
    int id = blockIdx.x * 256 + threadIdx.x;
    if (id >= BT_ * SO_) return;
    int j = id % SO_, b = id / SO_;
    float dcl = fminf(fmaxf(decay[j], 0.f), 15.f);
    float r   = expf(-dcl);
    float pp  = act[(size_t)b * D_ + il[j]] * act[(size_t)b * D_ + ir[j]];
    float a   = r * ao[id] + pp;
    float bb  = r * bo[id] + 1.f;
    ao[id] = a; bo[id] = bb;
    syncb[id] = (__bf16)(a * rsqrtf(bb));
}

// ---------------- log-softmax entropy + strided output writes -------------
__global__ void __launch_bounds__(256)
rowout(const float* __restrict__ predpre, float* __restrict__ out, int it) {
    __shared__ float red[256];
    int b = blockIdx.x, t = threadIdx.x;
    const float* pr = predpre + (size_t)b * D_;
    float v[3];
    float mx = -3.4e38f;
    #pragma unroll
    for (int i = 0; i < 3; ++i) { v[i] = pr[t + i * 256]; mx = fmaxf(mx, v[i]); }
    mx = blockReduceMax(mx, red);
    float se = 0.f;
    #pragma unroll
    for (int i = 0; i < 3; ++i) se += expf(v[i] - mx);
    se = blockReduceSum(se, red);
    float lse = mx + logf(se);
    float ei = 0.f;
    #pragma unroll
    for (int i = 0; i < 3; ++i) { float lp = v[i] - lse; ei += expf(lp) * lp; }
    ei = blockReduceSum(ei, red);
    float ne = (-ei) / logf((float)D_);
    size_t pb = (size_t)b * D_ * ITER_;
    #pragma unroll
    for (int i = 0; i < 3; ++i)
        out[pb + (size_t)(t + i * 256) * ITER_ + it] = v[i];
    if (t == 0) {
        size_t cb = (size_t)BT_ * D_ * ITER_ + (size_t)b * 2 * ITER_;
        out[cb + it]         = ne;
        out[cb + ITER_ + it] = 1.f - ne;
    }
}

// ---------------- host ----------------
extern "C" void kernel_launch(void* const* d_in, const int* in_sizes, int n_in,
                              void* d_out, int out_size, void* d_ws, size_t ws_size,
                              hipStream_t stream) {
    (void)in_sizes; (void)n_in; (void)out_size; (void)ws_size;
    const float* kv          = (const float*)d_in[0];
    // d_in[1], d_in[2] (action_idx_*) and q_w/q_b/decay_action are dead code.
    const int*   oil         = (const int*)d_in[3];
    const int*   oir         = (const int*)d_in[4];
    const float* proj1_w     = (const float*)d_in[5];
    const float* proj1_b     = (const float*)d_in[6];
    const float* proj1_g     = (const float*)d_in[7];
    const float* proj1_beta  = (const float*)d_in[8];
    const float* down_w      = (const float*)d_in[9];
    const float* down_b      = (const float*)d_in[10];
    const float* down_g      = (const float*)d_in[11];
    const float* down_beta   = (const float*)d_in[12];
    const float* up_w        = (const float*)d_in[13];
    const float* up_b        = (const float*)d_in[14];
    const float* up_g        = (const float*)d_in[15];
    const float* up_beta     = (const float*)d_in[16];
    const float* skip_g      = (const float*)d_in[17];
    const float* skip_beta   = (const float*)d_in[18];
    const float* nlm1_g      = (const float*)d_in[19];
    const float* nlm1_beta   = (const float*)d_in[20];
    const float* nlm1_w      = (const float*)d_in[21];
    const float* nlm1_b      = (const float*)d_in[22];
    const float* nlm1_T      = (const float*)d_in[23];
    const float* nlm2_g      = (const float*)d_in[24];
    const float* nlm2_beta   = (const float*)d_in[25];
    const float* nlm2_w      = (const float*)d_in[26];
    const float* nlm2_b      = (const float*)d_in[27];
    const float* nlm2_T      = (const float*)d_in[28];
    const float* o_w         = (const float*)d_in[31];
    const float* o_b         = (const float*)d_in[32];
    const float* decay_out   = (const float*)d_in[34];
    const float* start_act   = (const float*)d_in[35];
    const float* start_trace = (const float*)d_in[36];

    char* p = (char*)d_ws;
    auto carve = [&](size_t bytes) -> void* {
        void* r = (void*)p;
        p += (bytes + 255) & ~(size_t)255;
        return r;
    };
    float*  trace   = (float*)carve(sizeof(float) * M_ * BT_ * D_);   // ring buffer
    float*  act     = (float*)carve(sizeof(float) * BT_ * D_);
    float*  hpre    = (float*)carve(sizeof(float) * BT_ * D_);
    float*  predpre = (float*)carve(sizeof(float) * BT_ * D_);
    float*  ao      = (float*)carve(sizeof(float) * BT_ * SO_);
    float*  bo      = (float*)carve(sizeof(float) * BT_ * SO_);
    __bf16* xb      = (__bf16*)carve(sizeof(__bf16) * BT_ * K1_);
    __bf16* syncb   = (__bf16*)carve(sizeof(__bf16) * BT_ * SO_);
    __bf16* w1p     = (__bf16*)carve(sizeof(__bf16) * K1_ * D_);
    __bf16* owp     = (__bf16*)carve(sizeof(__bf16) * SO_ * D_);
    float*  out     = (float*)d_out;

    // one-time (per launch) setup: weight repack + state init
    repack_w<<<(K1_ * D_ + 255) / 256, 256, 0, stream>>>(proj1_w, w1p, K1_, D_);
    repack_w<<<(SO_ * D_ + 255) / 256, 256, 0, stream>>>(o_w, owp, SO_, D_);
    init_kv_act<<<(BT_ * D_) / 256, 256, 0, stream>>>(kv, start_act, act, xb);
    init_trace<<<(M_ * BT_ * D_) / 256, 256, 0, stream>>>(start_trace, trace);
    init_sync<<<(BT_ * SO_) / 256, 256, 0, stream>>>(start_act, oil, oir, ao, bo);

    dim3 ggrid(D_ / 32, BT_ / 64);
    for (int t = 0; t < ITER_; ++t) {
        gemm_bf16<<<ggrid, 128, 0, stream>>>(xb, w1p, proj1_b, hpre, K1_, D_);
        synnet_row<<<BT_, 256, 0, stream>>>(hpre, proj1_g, proj1_beta,
                                            down_w, down_b, down_g, down_beta,
                                            up_w, up_b, up_g, up_beta,
                                            skip_g, skip_beta, trace, t % M_);
        nlm_kernel<<<(BT_ * D_) / 256, 256, 0, stream>>>(trace, nlm1_g, nlm1_beta,
                                                         nlm1_w, nlm1_b, nlm1_T,
                                                         nlm2_g, nlm2_beta,
                                                         nlm2_w, nlm2_b, nlm2_T,
                                                         act, xb, t);
        outsync<<<(BT_ * SO_) / 256, 256, 0, stream>>>(act, oil, oir, decay_out,
                                                       ao, bo, syncb);
        gemm_bf16<<<ggrid, 128, 0, stream>>>(syncb, owp, o_b, predpre, SO_, D_);
        rowout<<<BT_, 256, 0, stream>>>(predpre, out, t);
    }
}
